// Merged_Attention_74388833566873
// MI455X (gfx1250) — compile-verified, matrix operand order checked by reading, not run on hardware
//
#include <hip/hip_runtime.h>
#include <hip/hip_fp16.h>

typedef __attribute__((ext_vector_type(16))) _Float16 v16h;
typedef __attribute__((ext_vector_type(8)))  _Float16 v8h;
typedef __attribute__((ext_vector_type(8)))  float    v8f;

#define HEADS 8
#define HDIM  64
#define NSEQ  2048
#define CDIM  512
#define SCALE 0.125f

// ---------------------------------------------------------------------------
// WMMA helper: D = A(16x32 f16) * B(32x16 f16) + C(16x16 f32)
// ---------------------------------------------------------------------------
__device__ __forceinline__ v8f wmma16(v16h a, v16h b, v8f c) {
  return __builtin_amdgcn_wmma_f32_16x16x32_f16(false, a, false, b,
                                                (short)0, c, false, false);
}

__device__ __forceinline__ v8f zero8() {
  v8f z;
#pragma unroll
  for (int i = 0; i < 8; ++i) z[i] = 0.0f;
  return z;
}

// ---------------------------------------------------------------------------
// DPP16 butterfly reductions within each 16-lane half (VALU-only, no DS pipe).
// xor1 = quad_perm(1,0,3,2)=0xB1 ; xor2 = quad_perm(2,3,0,1)=0x4E ;
// xor4 ~= row_half_mirror (0x141) ; xor8 ~= row_mirror (0x140)
// (mirror tricks valid since prior stages make groups of 4/8 uniform).
// ---------------------------------------------------------------------------
template <int CTRL>
__device__ __forceinline__ float dppf(float x) {
  const int i = __float_as_int(x);
  return __int_as_float(__builtin_amdgcn_update_dpp(i, i, CTRL, 0xf, 0xf, true));
}

__device__ __forceinline__ float red_max16(float x) {
  x = fmaxf(x, dppf<0xB1>(x));
  x = fmaxf(x, dppf<0x4E>(x));
  x = fmaxf(x, dppf<0x141>(x));
  x = fmaxf(x, dppf<0x140>(x));
  return x;
}

__device__ __forceinline__ float red_sum16(float x) {
  x += dppf<0xB1>(x);
  x += dppf<0x4E>(x);
  x += dppf<0x141>(x);
  x += dppf<0x140>(x);
  return x;
}

// A-fragment (16x32) layout per CDNA5 ISA: lane<16 holds K {kb..kb+7, kb+16..kb+23},
// lane>=16 holds the +8-shifted chunks.  Load from an f16 row (global or LDS).
__device__ __forceinline__ v16h load_a16(const _Float16* p, int kbase, int lane) {
  const int kb = kbase + ((lane & 16) >> 1);   // +8 halves for upper half-wave
  v8h lo = *reinterpret_cast<const v8h*>(p + kb);
  v8h hi = *reinterpret_cast<const v8h*>(p + kb + 16);
  v16h r;
#pragma unroll
  for (int i = 0; i < 8; ++i) { r[i] = lo[i]; r[8 + i] = hi[i]; }
  return r;
}

// Same fragment but sourced from an f32 row (convert on the fly).
__device__ __forceinline__ v16h load_a32(const float* row, int kbase, int lane) {
  const int kb = kbase + ((lane & 16) >> 1);
  float4 a0 = *reinterpret_cast<const float4*>(row + kb);
  float4 a1 = *reinterpret_cast<const float4*>(row + kb + 4);
  float4 a2 = *reinterpret_cast<const float4*>(row + kb + 16);
  float4 a3 = *reinterpret_cast<const float4*>(row + kb + 20);
  v16h r;
  r[0]  = (_Float16)a0.x; r[1]  = (_Float16)a0.y; r[2]  = (_Float16)a0.z; r[3]  = (_Float16)a0.w;
  r[4]  = (_Float16)a1.x; r[5]  = (_Float16)a1.y; r[6]  = (_Float16)a1.z; r[7]  = (_Float16)a1.w;
  r[8]  = (_Float16)a2.x; r[9]  = (_Float16)a2.y; r[10] = (_Float16)a2.z; r[11] = (_Float16)a2.w;
  r[12] = (_Float16)a3.x; r[13] = (_Float16)a3.y; r[14] = (_Float16)a3.z; r[15] = (_Float16)a3.w;
  return r;
}

// ---------------------------------------------------------------------------
// GEMM: C[M,512] = A[M,512] * W[512,512] + bias
//   OMODE 0: f16 output scattered to head-major [s][b][h][n][d]  (QKV)
//   OMODE 1: f32 output, row-major to d_out                      (out-proj)
// Block: 128 threads (4 waves). Block tile 64x64, wave strip 16x64.
// ---------------------------------------------------------------------------
template <typename AT, int OMODE>
__global__ __launch_bounds__(128) void gemm512_wmma(
    const AT* __restrict__ A, const float* __restrict__ W,
    const float* __restrict__ bias, _Float16* __restrict__ outH,
    float* __restrict__ outF, int mbase) {
  constexpr int K = CDIM, N = CDIM;
  const int tid  = threadIdx.x;
  const int lane = tid & 31;
  const int wave = tid >> 5;
  const int ln   = lane & 15;
  const int lh   = lane >> 4;
  const int n0   = blockIdx.x * 64;
  const int m0   = blockIdx.y * 64 + wave * 16;

  // Weight chunk, transposed to [n][k] f16; stride 40 halves (80 B, 16B aligned,
  // conflict-free b128 fragment reads).
  __shared__ __align__(16) _Float16 Bt[64 * 40];

  const AT* arow = A + (size_t)(m0 + ln) * K;

  v8f acc[4];
#pragma unroll
  for (int t = 0; t < 4; ++t) acc[t] = zero8();

  const int kl  = tid >> 2;        // 0..31  : k row within chunk
  const int ncl = (tid & 3) * 16;  // 0,16,32,48 : n start

  for (int k0 = 0; k0 < K; k0 += 32) {
    __syncthreads();
    {  // stage W[k0..k0+31][n0..n0+63] -> Bt[n][k] (f16)
      const float* bp = W + (size_t)(k0 + kl) * N + n0 + ncl;
      float tmp[16];
#pragma unroll
      for (int j = 0; j < 4; ++j) {
        float4 v = *reinterpret_cast<const float4*>(bp + j * 4);
        tmp[j * 4 + 0] = v.x; tmp[j * 4 + 1] = v.y;
        tmp[j * 4 + 2] = v.z; tmp[j * 4 + 3] = v.w;
      }
#pragma unroll
      for (int i = 0; i < 16; ++i)
        Bt[(ncl + i) * 40 + kl] = (_Float16)tmp[i];
      if (k0 + 32 < K) __builtin_prefetch(bp + 32 * N, 0, 0);
    }
    __syncthreads();

    v16h afrag;
    if constexpr (sizeof(AT) == 4) afrag = load_a32((const float*)arow, k0, lane);
    else                           afrag = load_a16((const _Float16*)arow, k0, lane);

#pragma unroll
    for (int t = 0; t < 4; ++t) {
      v16h bfrag = load_a16(&Bt[(t * 16 + ln) * 40], 0, lane);
      acc[t] = wmma16(afrag, bfrag, acc[t]);
    }
  }

  // Epilogue. C layout: VGPR r -> row r (lanes 0-15) / row r+8 (lanes 16-31),
  // col = lane%16.
#pragma unroll
  for (int t = 0; t < 4; ++t) {
    const int   c  = n0 + t * 16 + ln;
    const float bv = bias[c];
#pragma unroll
    for (int r = 0; r < 8; ++r) {
      const int   m   = mbase + m0 + r + 8 * lh;
      const float val = acc[t][r] + bv;
      if constexpr (OMODE == 0) {
        const int s = m >> 13, b = (m >> 11) & 3, n = m & 2047;
        const int h = c >> 6, d = c & 63;
        outH[((((size_t)s * 4 + b) * HEADS + h) * NSEQ + n) * HDIM + d] =
            (_Float16)val;
      } else {
        outF[(size_t)m * CDIM + c] = val;
      }
    }
  }
}

// ---------------------------------------------------------------------------
// Cross attention (flash-style, f16 WMMA, f32 accumulation).
// grid = (32 q-tiles, 8 heads, 8 = stream*4+batch); block = 128 (4 waves).
// Each wave owns 16 q rows; K tile row-major in LDS, V tile transposed.
// SCALE is folded into the Q fragments (2^-3 is exact in f16).
// ---------------------------------------------------------------------------
__global__ __launch_bounds__(128) void cross_attn_wmma(
    const _Float16* __restrict__ Qb, const _Float16* __restrict__ Kb,
    const _Float16* __restrict__ Vb, _Float16* __restrict__ Ob) {
  const int qt = blockIdx.x, h = blockIdx.y, sb = blockIdx.z;
  const int s = sb >> 2, b = sb & 3;
  const int tid = threadIdx.x;
  const int lane = tid & 31, wave = tid >> 5;
  const int ln = lane & 15, lh = lane >> 4;

  const size_t headQ  = (((size_t)s * 4 + b) * HEADS + h) * (size_t)(NSEQ * HDIM);
  const size_t headKV = (((size_t)(1 - s) * 4 + b) * HEADS + h) * (size_t)(NSEQ * HDIM);
  const _Float16* Qh = Qb + headQ;
  const _Float16* Kh = Kb + headKV;   // cross: K/V from the other stream
  const _Float16* Vh = Vb + headKV;

  __shared__ __align__(16) _Float16 Ks[64 * 72];    // [key][d], stride 72 halves
  __shared__ __align__(16) _Float16 Vt[64 * 72];    // [d][key], stride 72 halves
  __shared__ __align__(16) _Float16 Ps[4][16 * 40]; // per-wave P relayout buffer

  const int q0 = qt * 64 + wave * 16;
  const _Float16* qp = Qh + (size_t)(q0 + ln) * HDIM;
  v16h qf0 = load_a16(qp, 0, lane);
  v16h qf1 = load_a16(qp, 32, lane);
#pragma unroll
  for (int i = 0; i < 16; ++i) {   // fold softmax scale into Q (exact: 2^-3)
    qf0[i] = qf0[i] * (_Float16)SCALE;
    qf1[i] = qf1[i] * (_Float16)SCALE;
  }

  v8f oacc[4];
#pragma unroll
  for (int t = 0; t < 4; ++t) oacc[t] = zero8();
  float mi[8], li[8];
#pragma unroll
  for (int r = 0; r < 8; ++r) { mi[r] = -1e30f; li[r] = 0.0f; }

  const int key_c = tid >> 1;        // 0..63
  const int off_c = (tid & 1) * 32;  // 0 or 32 halves

  for (int kb0 = 0; kb0 < NSEQ; kb0 += 64) {
    __syncthreads();
    {  // cooperative staging: K row-major, V transposed
      const _Float16* kp = Kh + (size_t)(kb0 + key_c) * HDIM + off_c;
      _Float16* dk = &Ks[key_c * 72 + off_c];
#pragma unroll
      for (int j = 0; j < 4; ++j)
        *reinterpret_cast<v8h*>(dk + j * 8) =
            *reinterpret_cast<const v8h*>(kp + j * 8);

      const _Float16* vp = Vh + (size_t)(kb0 + key_c) * HDIM + off_c;
      v8h vv[4];
#pragma unroll
      for (int j = 0; j < 4; ++j)
        vv[j] = *reinterpret_cast<const v8h*>(vp + j * 8);
#pragma unroll
      for (int j = 0; j < 4; ++j)
#pragma unroll
        for (int i = 0; i < 8; ++i)
          Vt[(off_c + j * 8 + i) * 72 + key_c] = vv[j][i];

      if (kb0 + 64 < NSEQ) {
        __builtin_prefetch(kp + 64 * HDIM, 0, 0);
        __builtin_prefetch(vp + 64 * HDIM, 0, 0);
      }
    }
    __syncthreads();

#pragma unroll
    for (int pair = 0; pair < 2; ++pair) {  // 32 keys per pair
      v8f sA = zero8(), sB = zero8();
      {
        const int keyA = pair * 32 + ln;
        v16h bk0 = load_a16(&Ks[keyA * 72], 0, lane);
        v16h bk1 = load_a16(&Ks[keyA * 72], 32, lane);
        sA = wmma16(qf0, bk0, sA);
        sA = wmma16(qf1, bk1, sA);
        const int keyB = pair * 32 + 16 + ln;
        v16h bk2 = load_a16(&Ks[keyB * 72], 0, lane);
        v16h bk3 = load_a16(&Ks[keyB * 72], 32, lane);
        sB = wmma16(qf0, bk2, sB);
        sB = wmma16(qf1, bk3, sB);
      }
      // Online softmax: VALU-only DPP butterflies per 16-lane half.
      float alpha[8];
#pragma unroll
      for (int r = 0; r < 8; ++r) {
        const float va = sA[r];
        const float vb = sB[r];
        const float mx = red_max16(fmaxf(va, vb));
        const float mnew = fmaxf(mi[r], mx);
        const float pa = __expf(va - mnew);
        const float pb = __expf(vb - mnew);
        alpha[r] = __expf(mi[r] - mnew);
        const float rs = red_sum16(pa + pb);
        li[r] = li[r] * alpha[r] + rs;
        mi[r] = mnew;
        const int row = r + 8 * lh;   // P row (q), cols = keys of this pair
        Ps[wave][row * 40 + ln]      = (_Float16)pa;
        Ps[wave][row * 40 + 16 + ln] = (_Float16)pb;
      }
      // Cross-lane LDS relayout: wait for our own ds stores before re-reading.
      asm volatile("s_wait_dscnt 0" ::: "memory");
      const v16h pf = load_a16(&Ps[wave][ln * 40], 0, lane);
#pragma unroll
      for (int t = 0; t < 4; ++t) {
        v8f c;
#pragma unroll
        for (int r = 0; r < 8; ++r) c[r] = oacc[t][r] * alpha[r];
        v16h vf = load_a16(&Vt[(t * 16 + ln) * 72], pair * 32, lane);
        oacc[t] = wmma16(pf, vf, c);
      }
    }
  }

  // Normalize and store O (f16, row-major [ (s*4+b)*2048+n , h*64+d ]).
#pragma unroll
  for (int t = 0; t < 4; ++t) {
    const int c = h * HDIM + t * 16 + ln;
#pragma unroll
    for (int r = 0; r < 8; ++r) {
      const int n = q0 + r + 8 * lh;
      const float val = oacc[t][r] / li[r];
      Ob[((size_t)((s * 4 + b) * NSEQ + n)) * CDIM + c] = (_Float16)val;
    }
  }
}

// ---------------------------------------------------------------------------
extern "C" void kernel_launch(void* const* d_in, const int* in_sizes, int n_in,
                              void* d_out, int out_size, void* d_ws,
                              size_t ws_size, hipStream_t stream) {
  (void)in_sizes; (void)n_in; (void)out_size; (void)ws_size;
  const float* x_vis = (const float*)d_in[0];
  const float* x_ir  = (const float*)d_in[1];
  const float* Wq = (const float*)d_in[2];
  const float* bq = (const float*)d_in[3];
  const float* Wk = (const float*)d_in[4];
  const float* bk = (const float*)d_in[5];
  const float* Wv = (const float*)d_in[6];
  const float* bv = (const float*)d_in[7];
  const float* Wp = (const float*)d_in[8];
  const float* bp = (const float*)d_in[9];
  float* out = (float*)d_out;

  char* ws = (char*)d_ws;
  _Float16* Qb = (_Float16*)(ws);                             // 16 MB
  _Float16* Kb = (_Float16*)(ws + (size_t)16 * 1024 * 1024);  // 16 MB
  _Float16* Vb = (_Float16*)(ws + (size_t)32 * 1024 * 1024);  // 16 MB
  _Float16* Ob = (_Float16*)(ws + (size_t)48 * 1024 * 1024);  // 16 MB

  const dim3 blk(128);
  const dim3 gq(8, 128);  // 512/64 n-tiles x 8192/64 m-tiles

  // QKV projections, both streams stacked into head-major f16 buffers.
  gemm512_wmma<float, 0><<<gq, blk, 0, stream>>>(x_vis, Wq, bq, Qb, nullptr, 0);
  gemm512_wmma<float, 0><<<gq, blk, 0, stream>>>(x_ir,  Wq, bq, Qb, nullptr, 8192);
  gemm512_wmma<float, 0><<<gq, blk, 0, stream>>>(x_vis, Wk, bk, Kb, nullptr, 0);
  gemm512_wmma<float, 0><<<gq, blk, 0, stream>>>(x_ir,  Wk, bk, Kb, nullptr, 8192);
  gemm512_wmma<float, 0><<<gq, blk, 0, stream>>>(x_vis, Wv, bv, Vb, nullptr, 0);
  gemm512_wmma<float, 0><<<gq, blk, 0, stream>>>(x_ir,  Wv, bv, Vb, nullptr, 8192);

  // Cross attention: q_vis over k_ir/v_ir and q_ir over k_vis/v_vis.
  cross_attn_wmma<<<dim3(32, 8, 8), blk, 0, stream>>>(Qb, Kb, Vb, Ob);

  // Output projection straight into d_out (f32), rows already in tuple order.
  gemm512_wmma<_Float16, 1><<<dim3(8, 256), blk, 0, stream>>>(Ob, Wp, bp,
                                                              nullptr, out, 0);
}
